// GRU_53687091200305
// MI455X (gfx1250) — matrix operand order repaired and checked by reading, not compiled, not run on hardware
//
#include <hip/hip_runtime.h>
#include <stdint.h>
#include <math.h>

// Problem constants (N, T, D, H) = (16, 1024, 1024, 1024)
#define NB 16
#define TT 1024
#define DD 1024
#define HH 1024
#define H3 3072
#define NWG 64          // persistent workgroups for the scan
#define SCAN_THREADS 128

typedef __attribute__((ext_vector_type(16))) __bf16 v16bf;
typedef __attribute__((ext_vector_type(8)))  float  v8f;

// ---- CDNA5 WMMA 16x16x32 bf16 fragment index maps (ISA 7.12.2, wave32) ----
// A (16x32, MxK): lane l holds M=l%16; element e (0..15):
//   k = 16*(e>=8) + 8*(l>=16) + (e&7)
__device__ __forceinline__ int kA_of(int l, int e) {
  return ((e >> 3) << 4) + ((l >> 4) << 3) + (e & 7);
}
// B (32x16, KxN): lane l holds N=l%16; element e: k = 16*(e>=8) + 8*(l>=16) + (e&7)
__device__ __forceinline__ int kB_of(int l, int e) {
  return ((e >> 3) << 4) + ((l >> 4) << 3) + (e & 7);
}
// C/D (16x16 f32): lane l, vgpr v -> m = v + 8*(l>=16), n = l%16

__device__ __forceinline__ float sigmoidf_(float x) {
  return 1.0f / (1.0f + __expf(-x));
}

// -------------------------------------------------------------------------
// Init: h0 = carry[:,0,:]; zero grid-sync counters; write initial_h output.
// -------------------------------------------------------------------------
__global__ void gru_init_kernel(const float* __restrict__ carry,
                                const float* __restrict__ initial_h,
                                float* __restrict__ hbuf,
                                unsigned* __restrict__ syncv,
                                float* __restrict__ out_tail) {
  int idx = blockIdx.x * blockDim.x + threadIdx.x;
  if (idx < NB * HH) {
    int n = idx / HH, k = idx % HH;
    hbuf[idx] = carry[(size_t)n * TT * HH + k];   // carry[n][0][k]
  }
  if (idx < HH) out_tail[idx] = initial_h[idx];
  if (idx < 2) syncv[idx] = 0u;
}

// -------------------------------------------------------------------------
// gx = x @ w_i : (N*T, D) @ (D, 3H), bf16 WMMA, 64x64 tile per WG, 4 waves,
// each wave owns a 2x2 grid of 16x16 C tiles; K staged 32-wide through LDS
// in per-lane fragment order (f32 -> bf16 convert at pack time).
// -------------------------------------------------------------------------
__global__ void __launch_bounds__(128)
gru_gx_gemm(const float* __restrict__ x, const float* __restrict__ wi,
            float* __restrict__ gx) {
  __shared__ __align__(32) __bf16 sA[4 * 512];
  __shared__ __align__(32) __bf16 sB[4 * 512];

  const int tid  = threadIdx.x;
  const int lane = tid & 31;
  const int wv   = tid >> 5;
  const int bn   = blockIdx.x % (H3 / 64);
  const int bm   = blockIdx.x / (H3 / 64);
  const int wm   = wv & 1;        // m-tile pair 0/1
  const int wn   = wv >> 1;       // n-tile pair 0/1

  v8f acc00 = {}, acc01 = {}, acc10 = {}, acc11 = {};

  for (int kb = 0; kb < DD; kb += 32) {
    // pack A (64x32) and B (32x64) tiles into fragment-order LDS
    for (int i = tid; i < 2048; i += 128) {
      int t4 = i >> 9, w = i & 511, l = w >> 4, e = w & 15;
      int mn = l & 15;
      int row = bm * 64 + t4 * 16 + mn;
      sA[i] = (__bf16)x[(size_t)row * DD + kb + kA_of(l, e)];
      int col = bn * 64 + t4 * 16 + mn;
      sB[i] = (__bf16)wi[(size_t)(kb + kB_of(l, e)) * H3 + col];
    }
    __syncthreads();

    v16bf a0 = *(const v16bf*)&sA[(wm * 2 + 0) * 512 + lane * 16];
    v16bf a1 = *(const v16bf*)&sA[(wm * 2 + 1) * 512 + lane * 16];
    v16bf b0 = *(const v16bf*)&sB[(wn * 2 + 0) * 512 + lane * 16];
    v16bf b1 = *(const v16bf*)&sB[(wn * 2 + 1) * 512 + lane * 16];
    acc00 = __builtin_amdgcn_wmma_f32_16x16x32_bf16(false, a0, false, b0, (short)0, acc00, false, false);
    acc01 = __builtin_amdgcn_wmma_f32_16x16x32_bf16(false, a0, false, b1, (short)0, acc01, false, false);
    acc10 = __builtin_amdgcn_wmma_f32_16x16x32_bf16(false, a1, false, b0, (short)0, acc10, false, false);
    acc11 = __builtin_amdgcn_wmma_f32_16x16x32_bf16(false, a1, false, b1, (short)0, acc11, false, false);
    __syncthreads();
  }

  // epilogue: C layout m = v + 8*(lane>=16), n = lane%16
  const int mb = (lane >> 4) << 3;
  const int n  = lane & 15;
  v8f* accs[2][2] = {{&acc00, &acc01}, {&acc10, &acc11}};
  for (int i = 0; i < 2; ++i)
    for (int j = 0; j < 2; ++j) {
      v8f a = *accs[i][j];
      int row0 = bm * 64 + (wm * 2 + i) * 16;
      int col  = bn * 64 + (wn * 2 + j) * 16 + n;
#pragma unroll
      for (int v = 0; v < 8; ++v)
        gx[(size_t)(row0 + mb + v) * H3 + col] = a[v];
    }
}

// -------------------------------------------------------------------------
// Grid-wide sense-reversal barrier (persistent kernel, NWG workgroups).
// -------------------------------------------------------------------------
__device__ __forceinline__ void grid_sync(unsigned* cnt, unsigned* sense) {
  __syncthreads();
  if (threadIdx.x == 0) {
    __threadfence();
    volatile unsigned* vs = (volatile unsigned*)sense;
    unsigned s = *vs;
    if (atomicAdd(cnt, 1u) == (unsigned)(NWG - 1)) {
      *cnt = 0u;
      __threadfence();
      atomicAdd(sense, 1u);
    } else {
      while (*vs == s) __builtin_amdgcn_s_sleep(2);
    }
    __threadfence();
  }
  __syncthreads();
}

// -------------------------------------------------------------------------
// Persistent scan kernel. WG g owns zr columns [32g, 32g+32) and a columns
// [16g, 16g+16). Its w_h slice lives in LDS (bf16 B-fragments) for all T.
// Per step:
//   0) async-DMA hcur (16x1024 f32, 64KB) into LDS   [ASYNCcnt path]
//   A) z,r = sigmoid(gx_zr + h @ W_zr)   (2 col tiles x 2-way K split)
//      grid sync
//   B) a   = tanh(gx_a + (r*h) @ W_a)    (1 col tile x 4-way K split)
//      h'  = (1-z)*h + z*a  -> states out + double-buffered h; grid sync
// h_eff per step: reset[n,t] ? initial_h : h  (matches reference algebra).
// -------------------------------------------------------------------------
__global__ void __launch_bounds__(SCAN_THREADS)
gru_scan_kernel(const float* __restrict__ gx,
                const uint8_t* __restrict__ reset,
                const float* __restrict__ initial_h,
                const float* __restrict__ wh,
                const float* __restrict__ bias,
                float* __restrict__ hbuf,      // 2 * NB*HH (double buffer)
                float* __restrict__ zbuf,      // NB*HH
                float* __restrict__ rbuf,      // NB*HH
                unsigned* __restrict__ syncv,  // [cnt, sense]
                float* __restrict__ out) {
  extern __shared__ char smem_raw[];
  __bf16* wZR   = (__bf16*)smem_raw;          // 2 tiles * 32 kt * 512 (64KB)
  __bf16* wA    = wZR + 2 * 32 * 512;         // 1 tile  * 32 kt * 512 (32KB)
  __bf16* hF    = wA  + 32 * 512;             // h / r*h A-fragments   (32KB)
  float*  cS    = (float*)(hF + 32 * 512);    // 4 partial C tiles      (4KB)
  float*  hS    = cS + 4 * 256;               // staged hcur f32       (64KB)

  const int tid  = threadIdx.x;
  const int lane = tid & 31;
  const int wv   = tid >> 5;
  const int wg   = blockIdx.x;

  // LDS byte address of hS (flat addr low 32 bits == LDS offset per ISA 10.2)
  const unsigned hS_lds = (unsigned)(size_t)(void*)hS;

  // --- one-time: load this WG's w_h slice into LDS as bf16 B fragments ---
  for (int i = tid; i < 2 * 32 * 512; i += SCAN_THREADS) {
    int tile = i >> 14, rem = i & 16383, kt = rem >> 9, w = rem & 511;
    int l = w >> 4, e = w & 15;
    int k = kt * 32 + kB_of(l, e);
    int j = wg * 32 + tile * 16 + (l & 15);          // zr column
    wZR[i] = (__bf16)wh[(size_t)k * H3 + j];
  }
  for (int i = tid; i < 32 * 512; i += SCAN_THREADS) {
    int kt = i >> 9, w = i & 511, l = w >> 4, e = w & 15;
    int k = kt * 32 + kB_of(l, e);
    int j = 2 * HH + wg * 16 + (l & 15);             // a column
    wA[i] = (__bf16)wh[(size_t)k * H3 + j];
  }
  __syncthreads();

  const int cn  = lane & 15;           // C tile column within tile
  const int cmb = (lane >> 4) << 3;    // C tile row base (0 or 8)

  for (int t = 0; t < TT; ++t) {
    const float* hcur = hbuf + (size_t)(t & 1) * NB * HH;
    float*       hnxt = hbuf + (size_t)((t + 1) & 1) * NB * HH;

    // ---- stage hcur (64KB) into LDS via async DMA (ASYNCcnt-tracked) ----
    {
      const unsigned long long hsrc = (unsigned long long)(size_t)hcur;
#pragma unroll 8
      for (int q = 0; q < 32; ++q) {
        unsigned byteoff = (((unsigned)(wv * 32 + q) * 32u) + (unsigned)lane) * 16u;
        unsigned ldsoff  = hS_lds + byteoff;
        asm volatile("global_load_async_to_lds_b128 %0, %1, %2"
                     :: "v"(ldsoff), "v"(byteoff), "s"(hsrc) : "memory");
      }
      asm volatile("s_wait_asynccnt 0x0" ::: "memory");
    }
    if (t + 1 < TT) {  // warm next step's gx slice (global_prefetch_b8)
      __builtin_prefetch(&gx[((size_t)(tid & 15) * TT + (t + 1)) * H3 + wg * 32], 0, 3);
    }
    __syncthreads();

    // ---------------- Phase A: pack h_eff A-fragments ----------------
    for (int i = tid; i < 32 * 512; i += SCAN_THREADS) {
      int kt = i >> 9, w = i & 511, l = w >> 4, e = w & 15;
      int m = l & 15;
      int k = kt * 32 + kA_of(l, e);
      float hv = reset[m * TT + t] ? initial_h[k] : hS[m * HH + k];
      hF[i] = (__bf16)hv;
    }
    __syncthreads();

    {   // zr matmul: wave -> (col tile jt = wv>>1, K half = wv&1)
      int jt = wv >> 1, half = wv & 1;
      v8f acc = {};
#pragma unroll 4
      for (int kt = half * 16; kt < half * 16 + 16; ++kt) {
        v16bf a = *(const v16bf*)&hF[kt * 512 + lane * 16];
        v16bf b = *(const v16bf*)&wZR[jt * 16384 + kt * 512 + lane * 16];
        acc = __builtin_amdgcn_wmma_f32_16x16x32_bf16(false, a, false, b, (short)0, acc, false, false);
      }
#pragma unroll
      for (int v = 0; v < 8; ++v)
        cS[wv * 256 + (cmb + v) * 16 + cn] = acc[v];
    }
    __syncthreads();

    if (wv < 2) {   // reduce K halves + sigmoid; wave wv owns col tile wv
      int jt = wv;
      int j = wg * 32 + jt * 16 + cn;
#pragma unroll
      for (int v = 0; v < 8; ++v) {
        int m = cmb + v;
        int idx = m * 16 + cn;
        float s = cS[(2 * jt) * 256 + idx] + cS[(2 * jt + 1) * 256 + idx];
        float lin = gx[((size_t)m * TT + t) * H3 + j] + s + bias[j];
        float zr = sigmoidf_(lin);
        if (j < HH) zbuf[m * HH + j] = zr;
        else        rbuf[m * HH + (j - HH)] = zr;
      }
    }
    grid_sync(&syncv[0], &syncv[1]);

    // ---------------- Phase B: pack (r*h) A-fragments ----------------
    for (int i = tid; i < 32 * 512; i += SCAN_THREADS) {
      int kt = i >> 9, w = i & 511, l = w >> 4, e = w & 15;
      int m = l & 15;
      int k = kt * 32 + kA_of(l, e);
      float hv = reset[m * TT + t] ? initial_h[k] : hS[m * HH + k];
      hF[i] = (__bf16)(rbuf[m * HH + k] * hv);
    }
    __syncthreads();

    {   // a matmul: 4-way K split across waves
      v8f acc = {};
#pragma unroll 4
      for (int kt = wv * 8; kt < wv * 8 + 8; ++kt) {
        v16bf a = *(const v16bf*)&hF[kt * 512 + lane * 16];
        v16bf b = *(const v16bf*)&wA[kt * 512 + lane * 16];
        acc = __builtin_amdgcn_wmma_f32_16x16x32_bf16(false, a, false, b, (short)0, acc, false, false);
      }
#pragma unroll
      for (int v = 0; v < 8; ++v)
        cS[wv * 256 + (cmb + v) * 16 + cn] = acc[v];
    }
    __syncthreads();

    if (wv == 0) {  // reduce, tanh, h update, write states
      int ja = wg * 16 + cn;
#pragma unroll
      for (int v = 0; v < 8; ++v) {
        int m = cmb + v;
        int idx = m * 16 + cn;
        float s = cS[idx] + cS[256 + idx] + cS[512 + idx] + cS[768 + idx];
        float lin = gx[((size_t)m * TT + t) * H3 + 2 * HH + ja] + s + bias[2 * HH + ja];
        float av = tanhf(lin);
        float z  = zbuf[m * HH + ja];
        float hv = reset[m * TT + t] ? initial_h[ja] : hS[m * HH + ja];
        float hn = (1.0f - z) * hv + z * av;
        hnxt[m * HH + ja] = hn;
        size_t o = ((size_t)m * TT + t) * HH + ja;
        out[o] = hn;                               // states (copy 1)
        out[(size_t)NB * TT * HH + o] = hn;        // states (copy 2)
      }
    }
    grid_sync(&syncv[0], &syncv[1]);
  }
}

// -------------------------------------------------------------------------
extern "C" void kernel_launch(void* const* d_in, const int* in_sizes, int n_in,
                              void* d_out, int out_size, void* d_ws, size_t ws_size,
                              hipStream_t stream) {
  const float*   x         = (const float*)d_in[0];
  const uint8_t* reset     = (const uint8_t*)d_in[1];   // jax bool = 1 byte
  const float*   carry     = (const float*)d_in[2];
  const float*   initial_h = (const float*)d_in[3];
  const float*   w_i       = (const float*)d_in[4];
  const float*   w_h       = (const float*)d_in[5];
  const float*   b         = (const float*)d_in[6];
  float*         out       = (float*)d_out;

  // workspace layout
  char*     ws    = (char*)d_ws;
  unsigned* syncv = (unsigned*)ws;                         // 256 B reserved
  float*    hbuf  = (float*)(ws + 256);                    // 2 * NB*HH f32
  float*    zbuf  = hbuf + 2 * NB * HH;                    // NB*HH
  float*    rbuf  = zbuf + NB * HH;                        // NB*HH
  float*    gx    = rbuf + NB * HH;                        // NB*TT*H3 f32 (192 MB)

  gru_init_kernel<<<64, 256, 0, stream>>>(carry, initial_h, hbuf, syncv,
                                          out + 2ull * NB * TT * HH);

  gru_gx_gemm<<<(NB * TT / 64) * (H3 / 64), 128, 0, stream>>>(x, w_i, gx);

  size_t smem = (size_t)(2 * 32 * 512 + 32 * 512 + 32 * 512) * sizeof(__bf16)
              + (size_t)4 * 256 * sizeof(float)            // cS
              + (size_t)NB * HH * sizeof(float);           // hS stage = 200704 B
  hipFuncSetAttribute((const void*)gru_scan_kernel,
                      hipFuncAttributeMaxDynamicSharedMemorySize, (int)smem);
  gru_scan_kernel<<<NWG, SCAN_THREADS, smem, stream>>>(
      gx, reset, initial_h, w_h, b, hbuf, zbuf, rbuf, syncv, out);
}
